// self_attn_4629974745238
// MI455X (gfx1250) — matrix operand order repaired and checked by reading, not compiled, not run on hardware
//
#include <hip/hip_runtime.h>

// ---------------------------------------------------------------------------
// Self-attention forward for MI455X (gfx1250, wave32, WMMA).
//
// ~137 GFLOP of GEMM vs ~350 MB traffic -> matrix-core bound. All GEMMs run as
// f16 WMMA (v_wmma_f32_16x16x32_f16) with fp32 accumulate, NT-form so both
// operands are contiguous-in-K (fragments = 2x ds_load_b128 each).
// Data movement: double-buffered LDS tiles filled with CDNA5 async
// global->LDS loads (ASYNCcnt) when the toolchain exposes them; synchronous
// global->VGPR->ds_store fallback otherwise.
// ---------------------------------------------------------------------------

typedef __attribute__((ext_vector_type(16))) _Float16 v16h;
typedef __attribute__((ext_vector_type(8)))  _Float16 v8h;
typedef __attribute__((ext_vector_type(8)))  float    v8f;
typedef int v4i __attribute__((vector_size(16)));   // builtin param type

#define DIMD 1024
#define SEQ  2048
#define NBAT 4

#if defined(__AMDGCN__) && __has_builtin(__builtin_amdgcn_global_load_async_to_lds_b128)
#define CDNA5_ASYNC 1
#else
#define CDNA5_ASYNC 0
#endif

__device__ __forceinline__ void wait_async0() {
#if defined(__AMDGCN__) && __has_builtin(__builtin_amdgcn_s_wait_asynccnt)
  __builtin_amdgcn_s_wait_asynccnt(0);
#elif defined(__AMDGCN__)
  asm volatile("s_wait_asynccnt 0x0" ::: "memory");
#endif
}

// ---------------- conversion kernels ----------------

__global__ __launch_bounds__(256)
void k_cvt_f16(const float* __restrict__ in, _Float16* __restrict__ out, int n) {
  int i = blockIdx.x * 256 + threadIdx.x;
  if (i < n) out[i] = (_Float16)in[i];
}

__global__ __launch_bounds__(256)
void k_transpose_cvt(const float* __restrict__ in, _Float16* __restrict__ out,
                     int rows, int cols) {
  int i = blockIdx.x * 256 + threadIdx.x;
  if (i < rows * cols) {
    int r = i / cols, c = i % cols;
    out[(size_t)c * rows + r] = (_Float16)in[i];
  }
}

// ---------------- generic NT WMMA GEMM ----------------
// 256 threads = 8 waves. Block tile 128x128, wave tile 64x32 (4x2 WMMA tiles).
// MODE 0: qkv projection (+bqkv), split into q[s][d], k[s][d], v^T[d][s] (f16)
// MODE 1: scores = q k^T * (1/32); mask==0 -> 1e-10; store fp32
// MODE 2: attn @ V -> f16 [s][d]
// MODE 3: out projection (+bout) -> fp32 d_out

template <int MODE>
__global__ __launch_bounds__(256) void k_gemm_nt(
    const _Float16* __restrict__ A, const _Float16* __restrict__ B,
    int lda, int ldb, int K,
    long long aBatch, long long bBatch,
    const float* __restrict__ bias, const int* __restrict__ mask,
    void* __restrict__ o0, void* __restrict__ o1, void* __restrict__ o2)
{
  // 32 cols + 8 pad per row (80B rows, 16B aligned); two buffers for pipeline
  __shared__ _Float16 As[2][128 * 40];
  __shared__ _Float16 Bs[2][128 * 40];

  const int z = blockIdx.z;
  A += (size_t)z * aBatch;
  B += (size_t)z * bBatch;

  const int blockM = blockIdx.x * 128;
  const int blockN = blockIdx.y * 128;

  const int t      = threadIdx.x;
  const int lane   = t & 31;
  const int waveId = t >> 5;
  const int wm     = (waveId >> 2) * 64;   // wave row offset (0 / 64)
  const int wn     = (waveId & 3) * 32;    // wave col offset (0..96)
  const int lhi    = lane >> 4;            // half-wave select
  const int llo    = lane & 15;

  // cooperative tile loads: thread -> (row, 16-element half-row) = 2x 16B
  const int loadRow = t >> 1;
  const int loadCol = (t & 1) * 16;
  const _Float16* Ag = A + (size_t)(blockM + loadRow) * lda + loadCol;
  const _Float16* Bg = B + (size_t)(blockN + loadRow) * ldb + loadCol;
  const int ldsOff = loadRow * 40 + loadCol;

  const v8f vzero = {};
  v8f acc[4][2];
#pragma unroll
  for (int mt = 0; mt < 4; ++mt)
#pragma unroll
    for (int nt = 0; nt < 2; ++nt) acc[mt][nt] = vzero;

#if CDNA5_ASYNC
  // prologue: async-fill buffer 0 with the first K-chunk
  {
    __builtin_amdgcn_global_load_async_to_lds_b128(
        (v4i*)(Ag), (v4i*)&As[0][ldsOff], 0, 0);
    __builtin_amdgcn_global_load_async_to_lds_b128(
        (v4i*)(Ag + 8), (v4i*)&As[0][ldsOff + 8], 0, 0);
    __builtin_amdgcn_global_load_async_to_lds_b128(
        (v4i*)(Bg), (v4i*)&Bs[0][ldsOff], 0, 0);
    __builtin_amdgcn_global_load_async_to_lds_b128(
        (v4i*)(Bg + 8), (v4i*)&Bs[0][ldsOff + 8], 0, 0);
  }
  int cur = 0;
  for (int kk = 0; kk < K; kk += 32) {
    wait_async0();        // this wave's async writes to buf[cur] are in LDS
    __syncthreads();      // everyone's are
    if (kk + 32 < K) {    // prefetch next chunk into the other buffer
      const int nb = cur ^ 1;
      __builtin_amdgcn_global_load_async_to_lds_b128(
          (v4i*)(Ag + kk + 32), (v4i*)&As[nb][ldsOff], 0, 0);
      __builtin_amdgcn_global_load_async_to_lds_b128(
          (v4i*)(Ag + kk + 40), (v4i*)&As[nb][ldsOff + 8], 0, 0);
      __builtin_amdgcn_global_load_async_to_lds_b128(
          (v4i*)(Bg + kk + 32), (v4i*)&Bs[nb][ldsOff], 0, 0);
      __builtin_amdgcn_global_load_async_to_lds_b128(
          (v4i*)(Bg + kk + 40), (v4i*)&Bs[nb][ldsOff + 8], 0, 0);
    }
#else
  int cur = 0;
  for (int kk = 0; kk < K; kk += 32) {
    v8h ga0 = *(const v8h*)(Ag + kk);
    v8h ga1 = *(const v8h*)(Ag + kk + 8);
    v8h gb0 = *(const v8h*)(Bg + kk);
    v8h gb1 = *(const v8h*)(Bg + kk + 8);
    __syncthreads();   // previous iteration's LDS reads complete
    *(v8h*)&As[0][ldsOff]     = ga0;
    *(v8h*)&As[0][ldsOff + 8] = ga1;
    *(v8h*)&Bs[0][ldsOff]     = gb0;
    *(v8h*)&Bs[0][ldsOff + 8] = gb1;
    __syncthreads();
#endif

    // B fragments: lane l -> col N=llo, K = lhi*16 + i  (16 contiguous f16)
    v16h bf[2];
#pragma unroll
    for (int nt = 0; nt < 2; ++nt) {
      const _Float16* p = &Bs[cur][(wn + nt * 16 + llo) * 40 + lhi * 16];
      v8h b0 = *(const v8h*)p;
      v8h b1 = *(const v8h*)(p + 8);
      v16h f;
#pragma unroll
      for (int i = 0; i < 8; ++i) { f[i] = b0[i]; f[i + 8] = b1[i]; }
      bf[nt] = f;
    }

    // A fragments: lane l -> row M=llo, K = lhi*16 + i  (16 contiguous f16)
#pragma unroll
    for (int mt = 0; mt < 4; ++mt) {
      const _Float16* p = &As[cur][(wm + mt * 16 + llo) * 40 + lhi * 16];
      v8h a0 = *(const v8h*)p;
      v8h a1 = *(const v8h*)(p + 8);
      v16h af;
#pragma unroll
      for (int i = 0; i < 8; ++i) { af[i] = a0[i]; af[i + 8] = a1[i]; }

      acc[mt][0] = __builtin_amdgcn_wmma_f32_16x16x32_f16(
          false, af, false, bf[0], (short)0, acc[mt][0], false, false);
      acc[mt][1] = __builtin_amdgcn_wmma_f32_16x16x32_f16(
          false, af, false, bf[1], (short)0, acc[mt][1], false, false);
    }
#if CDNA5_ASYNC
    cur ^= 1;
#endif
  }

  // Epilogue. C layout: VGPR r, lane l -> M = r + lhi*8, N = llo.
#pragma unroll
  for (int mt = 0; mt < 4; ++mt) {
#pragma unroll
    for (int nt = 0; nt < 2; ++nt) {
#pragma unroll
      for (int r = 0; r < 8; ++r) {
        const int row = blockM + wm + mt * 16 + lhi * 8 + r;
        const int col = blockN + wn + nt * 16 + llo;
        float v = acc[mt][nt][r];

        if (MODE == 0) {              // qkv projection
          v += bias[col];
          const int seg = col >> 10, d = col & 1023;
          const int b = row >> 11, s = row & 2047;
          const _Float16 hv = (_Float16)v;
          if (seg == 0)
            ((_Float16*)o0)[((size_t)b * SEQ + s) * DIMD + d] = hv;
          else if (seg == 1)
            ((_Float16*)o1)[((size_t)b * SEQ + s) * DIMD + d] = hv;
          else
            ((_Float16*)o2)[((size_t)b * DIMD + d) * SEQ + s] = hv;  // v^T
        } else if (MODE == 1) {       // scores + scale + mask
          v *= 0.03125f;              // 1/sqrt(1024)
          const size_t idx = ((size_t)z * SEQ + row) * SEQ + col;
          ((float*)o0)[idx] = (mask[idx] == 0) ? 1e-10f : v;
        } else if (MODE == 2) {       // attn @ V
          ((_Float16*)o0)[((size_t)(z * SEQ + row)) * DIMD + col] = (_Float16)v;
        } else {                      // output projection
          ((float*)o0)[(size_t)row * DIMD + col] = v + bias[col];
        }
      }
    }
  }
}

// ---------------- row softmax (2048 floats per row) ----------------

__global__ __launch_bounds__(256)
void k_softmax_row(const float* __restrict__ sc, _Float16* __restrict__ attn) {
  __shared__ float red[256];
  const size_t base = (size_t)blockIdx.x * SEQ;
  const int t = threadIdx.x;

  float v[8];
  float mx = -3.4e38f;
#pragma unroll
  for (int j = 0; j < 8; ++j) {
    v[j] = sc[base + t + j * 256];
    mx = fmaxf(mx, v[j]);
  }
  red[t] = mx;
  __syncthreads();
  for (int off = 128; off > 0; off >>= 1) {
    if (t < off) red[t] = fmaxf(red[t], red[t + off]);
    __syncthreads();
  }
  mx = red[0];
  __syncthreads();

  float s = 0.f;
#pragma unroll
  for (int j = 0; j < 8; ++j) {
    v[j] = __expf(v[j] - mx);
    s += v[j];
  }
  red[t] = s;
  __syncthreads();
  for (int off = 128; off > 0; off >>= 1) {
    if (t < off) red[t] += red[t + off];
    __syncthreads();
  }
  const float inv = 1.0f / red[0];
#pragma unroll
  for (int j = 0; j < 8; ++j)
    attn[base + t + j * 256] = (_Float16)(v[j] * inv);
}

// ---------------- launcher ----------------

extern "C" void kernel_launch(void* const* d_in, const int* in_sizes, int n_in,
                              void* d_out, int out_size, void* d_ws, size_t ws_size,
                              hipStream_t stream) {
  (void)in_sizes; (void)n_in; (void)out_size; (void)ws_size;

  const float* x    = (const float*)d_in[0];   // [4,2048,1024]
  const int*   mask = (const int*)d_in[1];     // [4,2048,2048]
  const float* Wqkv = (const float*)d_in[2];   // [1024,3072]
  const float* bqkv = (const float*)d_in[3];   // [3072]
  const float* Wout = (const float*)d_in[4];   // [1024,1024]
  const float* bout = (const float*)d_in[5];   // [1024]
  float* out = (float*)d_out;                  // [4,2048,1024] fp32

  // workspace layout (bytes); total 142,606,336 (~136 MB)
  char* ws = (char*)d_ws;
  _Float16* x_h     = (_Float16*)(ws + 0);          // 16MB, reused as attn_out
  _Float16* wqkv_t  = (_Float16*)(ws + 16777216);   // 6MB  [3072][1024]
  _Float16* wout_t  = (_Float16*)(ws + 23068672);   // 2MB  [1024][1024]
  _Float16* q_h     = (_Float16*)(ws + 25165824);   // 16MB [b][s][d]
  _Float16* k_h     = (_Float16*)(ws + 41943040);   // 16MB [b][s][d]
  _Float16* v_t     = (_Float16*)(ws + 58720256);   // 16MB [b][d][s]
  float*    scores  = (float*)  (ws + 75497472);    // 64MB [b][s][s]
  _Float16* attn    = q_h;       // 32MB reuse of q_h+k_h (dead after scores)
  _Float16* attn_o  = x_h;       // 16MB reuse of x_h (dead after GEMM1)

  // 1) precision conversion / transposes
  k_cvt_f16<<<(NBAT * SEQ * DIMD) / 256, 256, 0, stream>>>(x, x_h, NBAT * SEQ * DIMD);
  k_transpose_cvt<<<(DIMD * 3 * DIMD) / 256, 256, 0, stream>>>(Wqkv, wqkv_t, DIMD, 3 * DIMD);
  k_transpose_cvt<<<(DIMD * DIMD) / 256, 256, 0, stream>>>(Wout, wout_t, DIMD, DIMD);

  // 2) qkv = x @ Wqkv + bqkv   (M=8192, N=3072, K=1024)
  {
    dim3 g(8192 / 128, 3072 / 128, 1);
    k_gemm_nt<0><<<g, 256, 0, stream>>>(x_h, wqkv_t, DIMD, DIMD, DIMD, 0, 0,
                                        bqkv, nullptr, q_h, k_h, v_t);
  }

  // 3) scores = q k^T / 32, mask -> 1e-10   (per batch M=N=2048, K=1024)
  {
    dim3 g(SEQ / 128, SEQ / 128, NBAT);
    k_gemm_nt<1><<<g, 256, 0, stream>>>(q_h, k_h, DIMD, DIMD, DIMD,
                                        (long long)SEQ * DIMD, (long long)SEQ * DIMD,
                                        nullptr, mask, scores, nullptr, nullptr);
  }

  // 4) softmax rows -> f16 attn
  k_softmax_row<<<NBAT * SEQ, 256, 0, stream>>>(scores, attn);

  // 5) attn @ V   (per batch M=2048, N=1024, K=2048)
  {
    dim3 g(SEQ / 128, DIMD / 128, NBAT);
    k_gemm_nt<2><<<g, 256, 0, stream>>>(attn, v_t, SEQ, SEQ, SEQ,
                                        (long long)SEQ * SEQ, (long long)DIMD * SEQ,
                                        nullptr, nullptr, attn_o, nullptr, nullptr);
  }

  // 6) out = attn_out @ Wout + bout   (M=8192, N=1024, K=1024)
  {
    dim3 g(8192 / 128, DIMD / 128, 1);
    k_gemm_nt<3><<<g, 256, 0, stream>>>(attn_o, wout_t, DIMD, DIMD, DIMD, 0, 0,
                                        bout, nullptr, out, nullptr, nullptr);
  }
}
